// MixFormer_82944408420687
// MI455X (gfx1250) — matrix operand order repaired and checked by reading, not compiled, not run on hardware
//
#include <hip/hip_runtime.h>
#include <hip/hip_bf16.h>

typedef __attribute__((ext_vector_type(16))) _Float16 v16h;
typedef __attribute__((ext_vector_type(8)))  float    v8f;

constexpr int   MTOK   = 100352;           // 2048 windows * 49 tokens
constexpr float BN_RS  = 0.9999950000374997f;   // 1/sqrt(1+1e-5)

// ---- workspace layout (bytes) ----
constexpr size_t OFF_W16  = 0;                         // f16 weights (<=0.5MB)
constexpr size_t OFF_POOL = 524288;                    // 32*256 f32
constexpr size_t OFF_GATE = 557056;                    // 32*128 f32
constexpr size_t OFF_BIAS = 573440;                    // 8*64*64 f32 attention bias (padded)
constexpr size_t OFF_R1   = 704512;                    // TC f32 M*256 -> QKV f32 M*384
constexpr size_t OFF_R2   = OFF_R1 + 154140672ull;     // XCNN f32 NCHW -> CONCAT f16 M*256
constexpr size_t OFF_R3   = OFF_R2 + 102760448ull;     // CONV f32 NCHW -> AOUT f32 M*128
constexpr size_t OFF_R4   = OFF_R3 + 102760448ull;     // XCH f16 NHWC
constexpr size_t OFF_R5   = OFF_R4 + 51380224ull;      // TA f32 M*128 -> CPROJ f32 M*128
constexpr size_t OFF_R6   = OFF_R5 + 51380224ull;      // XATTH f16 M*128

__device__ __forceinline__ float gelu_f(float v) {
    return 0.5f * v * (1.0f + erff(v * 0.70710678118654752f));
}
__device__ __forceinline__ float sigmoid_f(float v) {
    return 1.0f / (1.0f + __expf(-v));
}

// ---------------- f32 -> f16 copy ----------------
__global__ void cvt_f16_k(const float* __restrict__ in, _Float16* __restrict__ out, size_t n) {
    size_t i = (size_t)blockIdx.x * 256 + threadIdx.x;
    if (i < n) out[i] = (_Float16)in[i];
}

// ---------------- precompute padded attention bias: [head][qn 64][kn 64] ----------------
__global__ void bias_k(const float* __restrict__ rpb, float* __restrict__ bias) {
    int idx = blockIdx.x * 256 + threadIdx.x;   // 8*64*64 = 32768
    int kn = idx & 63, t = idx >> 6;
    int qn = t & 63, head = t >> 6;
    float v = -1e30f;
    if (qn < 49 && kn < 49) {
        int dy = qn / 7 - kn / 7 + 6, dx = qn % 7 - kn % 7 + 6;
        v = rpb[(dy * 13 + dx) * 8 + head];
    }
    bias[idx] = v;
}

// ---------------- WMMA GEMM: C = A(MxK) @ W(KxN) + bias ----------------
// A: f32 or f16 row-major; W: f16 row-major; C: f32 row-major.
// block = 128 threads (4 waves); tile 64x64; wave w -> rows [16w,16w+16), 4 col-tiles.
template <typename AT>
__global__ __launch_bounds__(128) void gemm_wmma_k(
    const AT* __restrict__ A, const _Float16* __restrict__ W,
    const float* __restrict__ bias, float* __restrict__ Cout,
    int M, int K, int N)
{
    __shared__ _Float16 As[64][40];   // 64 rows x 32 K (padded, rows 80B aligned)
    __shared__ _Float16 Ws[32][72];   // 32 K x 64 cols (padded, rows 144B aligned)
    const int tid  = threadIdx.x;
    const int wv   = tid >> 5;
    const int lane = tid & 31;
    const int lh   = lane & 15;
    const int hi   = lane >> 4;
    const int row0 = blockIdx.x * 64;
    const int col0 = blockIdx.y * 64;

    v8f acc[4] = {{}, {}, {}, {}};

    for (int k0 = 0; k0 < K; k0 += 32) {
        __syncthreads();
        {   // A tile: thread t loads 16 elems: row = t>>1, seg = t&1
            int r = tid >> 1, s = tid & 1;
            if constexpr (sizeof(AT) == 4) {
                const float* src = (const float*)A + (size_t)(row0 + r) * K + k0 + 16 * s;
#pragma unroll
                for (int i = 0; i < 16; ++i) As[r][16 * s + i] = (_Float16)src[i];
            } else {
                const uint4* src = (const uint4*)((const _Float16*)A + (size_t)(row0 + r) * K + k0 + 16 * s);
                uint4* dst = (uint4*)&As[r][16 * s];
                dst[0] = src[0]; dst[1] = src[1];
            }
        }
        {   // W tile: thread t loads 16 halfs: r = t>>2 (K row), seg = t&3
            int r = tid >> 2, s = tid & 3;
            const uint4* src = (const uint4*)(W + (size_t)(k0 + r) * N + col0 + 16 * s);
            uint4* dst = (uint4*)&Ws[r][16 * s];
            dst[0] = src[0]; dst[1] = src[1];
        }
        __syncthreads();

        // A fragment (ISA 16-bit A 16x32 layout)
        v16h af;
#pragma unroll
        for (int e = 0; e < 16; ++e) {
            int j = e >> 1, h = e & 1;
            int kk = ((j >> 2) << 4) + ((j & 3) << 1) + h + (hi << 3);
            af[e] = As[(wv << 4) + lh][kk];
        }
#pragma unroll
        for (int jt = 0; jt < 4; ++jt) {
            v16h bf;   // B fragment: lane = N col, halves 0..15 = K (table order: hi*8 + e, +8 for e>=8)
#pragma unroll
            for (int e = 0; e < 16; ++e) bf[e] = Ws[(hi << 3) + e + (e & 8)][(jt << 4) + lh];
            acc[jt] = __builtin_amdgcn_wmma_f32_16x16x32_f16(
                false, af, false, bf, (short)0, acc[jt], false, false);
        }
    }
#pragma unroll
    for (int jt = 0; jt < 4; ++jt) {
        int n = col0 + (jt << 4) + lh;
        float bv = bias ? bias[n] : 0.0f;
#pragma unroll
        for (int r = 0; r < 8; ++r) {
            int m = row0 + (wv << 4) + (hi << 3) + r;    // C/D layout
            Cout[(size_t)m * N + n] = acc[jt][r] + bv;
        }
    }
}

// ---------------- LayerNorm: wave per row ----------------
__global__ __launch_bounds__(128) void ln_k(
    const float* __restrict__ in, const float* __restrict__ g, const float* __restrict__ b,
    float* __restrict__ out32, _Float16* __restrict__ out16, int C, int ostride)
{
    int wv = threadIdx.x >> 5, lane = threadIdx.x & 31;
    int row = blockIdx.x * 4 + wv;
    const float* p = in + (size_t)row * C;
    float v[8];
    int n = C >> 5;
    float s = 0.f;
    for (int i = 0; i < n; ++i) { v[i] = p[lane + 32 * i]; s += v[i]; }
    for (int off = 16; off >= 1; off >>= 1) s += __shfl_xor(s, off, 32);
    float mean = s / (float)C;
    float vs = 0.f;
    for (int i = 0; i < n; ++i) { float d = v[i] - mean; vs += d * d; }
    for (int off = 16; off >= 1; off >>= 1) vs += __shfl_xor(vs, off, 32);
    float rstd = rsqrtf(vs / (float)C + 1e-5f);
    for (int i = 0; i < n; ++i) {
        int c = lane + 32 * i;
        float o = (v[i] - mean) * rstd * g[c] + b[c];
        if (out16) out16[(size_t)row * ostride + c] = (_Float16)o;
        else       out32[(size_t)row * ostride + c] = o;
    }
}

// ---------------- window_reverse: (B_,49,C) tokens -> (B,C,56,56) ----------------
__global__ void win_rev_k(const float* __restrict__ tok, float* __restrict__ img) {
    size_t idx = (size_t)blockIdx.x * 256 + threadIdx.x;   // over 32*256*3136
    int x = idx % 56; size_t t = idx / 56;
    int y = t % 56;  t /= 56;
    int c = t % 256; int b = (int)(t / 256);
    int w = b * 64 + (y / 7) * 8 + (x / 7);
    int n = (y % 7) * 7 + (x % 7);
    img[idx] = tok[((size_t)w * 49 + n) * 256 + c];
}

// ---------------- depthwise 3x3 + bias + BN(eval) + GELU ----------------
__global__ void dwconv_k(const float* __restrict__ in, const float* __restrict__ wgt,
                         const float* __restrict__ bias, const float* __restrict__ bng,
                         const float* __restrict__ bnb, float* __restrict__ out) {
    size_t idx = (size_t)blockIdx.x * 256 + threadIdx.x;   // over 32*256*3136
    int x = idx % 56; size_t t = idx / 56;
    int y = t % 56;  t /= 56;
    int c = t % 256; int b = (int)(t / 256);
    const float* base = in + ((size_t)b * 256 + c) * 3136;
    float s = 0.f;
#pragma unroll
    for (int ky = 0; ky < 3; ++ky) {
#pragma unroll
        for (int kx = 0; kx < 3; ++kx) {
            int yy = y + ky - 1, xx = x + kx - 1;
            if (yy >= 0 && yy < 56 && xx >= 0 && xx < 56)
                s += base[yy * 56 + xx] * wgt[c * 9 + ky * 3 + kx];
        }
    }
    s += bias[c];
    s = s * (bng[c] * BN_RS) + bnb[c];
    out[idx] = gelu_f(s);
}

// ---------------- GAP over HxW per (b,c) ----------------
__global__ __launch_bounds__(256) void pool_k(const float* __restrict__ in, float* __restrict__ out) {
    int bc = blockIdx.x;
    const float* p = in + (size_t)bc * 3136;
    float s = 0.f;
    for (int i = threadIdx.x; i < 3136; i += 256) s += p[i];
    __shared__ float sm[256];
    sm[threadIdx.x] = s; __syncthreads();
    for (int o = 128; o >= 1; o >>= 1) {
        if ((int)threadIdx.x < o) sm[threadIdx.x] += sm[threadIdx.x + o];
        __syncthreads();
    }
    if (threadIdx.x == 0) out[bc] = sm[0] * (1.0f / 3136.0f);
}

// ---------------- channel interaction MLP -> sigmoid gate (B,128) ----------------
__global__ __launch_bounds__(128) void ci_k(const float* __restrict__ pool,
    const float* __restrict__ w1, const float* __restrict__ b1,
    const float* __restrict__ bng, const float* __restrict__ bnb,
    const float* __restrict__ w2, const float* __restrict__ b2, float* __restrict__ gate)
{
    int b = blockIdx.x, t = threadIdx.x;
    __shared__ float h[32];
    if (t < 32) {
        float s = b1[t];
        for (int c = 0; c < 256; ++c) s += pool[b * 256 + c] * w1[c * 32 + t];
        s = s * (bng[t] * BN_RS) + bnb[t];
        h[t] = gelu_f(s);
    }
    __syncthreads();
    float s = b2[t];
    for (int j = 0; j < 32; ++j) s += h[j] * w2[j * 128 + t];
    gate[b * 128 + t] = sigmoid_f(s);
}

// ---------------- (B,C,H,W) f32 -> (B,HW,C) f16 ----------------
__global__ void to_nhwc_h_k(const float* __restrict__ in, _Float16* __restrict__ out) {
    size_t idx = (size_t)blockIdx.x * 256 + threadIdx.x;   // over 32*3136*256 (out layout)
    int c = idx % 256; size_t t = idx / 256;
    int hw = t % 3136; int b = (int)(t / 3136);
    out[idx] = (_Float16)in[((size_t)b * 256 + c) * 3136 + hw];
}

// ---------------- windowed attention, one block per (window, head) ----------------
__global__ __launch_bounds__(128) void attn_k(const float* __restrict__ qkv,
    const float* __restrict__ gate, const float* __restrict__ biasbuf, float* __restrict__ aout)
{
    int blk = blockIdx.x;
    int w = blk >> 3, head = blk & 7;
    int bidx = w >> 6;
    __shared__ _Float16 Qs[64][34], Ks[64][34], Vs[64][34];
    __shared__ float    S[64][65];
    __shared__ _Float16 P[64][72];
    int tid = threadIdx.x;

    // stage Q/K/V with b128 loads: 64 rows x 4 float4 groups
    for (int i = tid; i < 256; i += 128) {
        int n = i >> 2, dg = (i & 3) << 2;
        float4 q = make_float4(0.f, 0.f, 0.f, 0.f), k = q, v = q;
        if (n < 49) {
            const float4* base = (const float4*)(qkv + ((size_t)w * 49 + n) * 384 + head * 16 + dg);
            q = base[0]; k = base[32]; v = base[64];       // +128 / +256 floats
            float4 g = *(const float4*)(gate + bidx * 128 + head * 16 + dg);
            q.x *= 0.25f; q.y *= 0.25f; q.z *= 0.25f; q.w *= 0.25f;   // hd^-0.5
            v.x *= g.x;   v.y *= g.y;   v.z *= g.z;   v.w *= g.w;
        }
        Qs[n][dg + 0] = (_Float16)q.x; Qs[n][dg + 1] = (_Float16)q.y;
        Qs[n][dg + 2] = (_Float16)q.z; Qs[n][dg + 3] = (_Float16)q.w;
        Ks[n][dg + 0] = (_Float16)k.x; Ks[n][dg + 1] = (_Float16)k.y;
        Ks[n][dg + 2] = (_Float16)k.z; Ks[n][dg + 3] = (_Float16)k.w;
        Vs[n][dg + 0] = (_Float16)v.x; Vs[n][dg + 1] = (_Float16)v.y;
        Vs[n][dg + 2] = (_Float16)v.z; Vs[n][dg + 3] = (_Float16)v.w;
#pragma unroll
        for (int j = 0; j < 4; ++j) {   // zero the K-padding half [16,32)
            Qs[n][16 + dg + j] = (_Float16)0.f;
            Ks[n][16 + dg + j] = (_Float16)0.f;
            Vs[n][16 + dg + j] = (_Float16)0.f;
        }
    }
    __syncthreads();

    int wv = tid >> 5, lane = tid & 31, lh = lane & 15, hi = lane >> 4;
    const float* bh = biasbuf + head * 4096;    // padded bias tile (-1e30 outside 49x49)

    // ---- Q @ K^T (wave wv handles query rows [16wv,16wv+16)) ----
    v16h af;
#pragma unroll
    for (int e = 0; e < 16; ++e) {
        int j = e >> 1, h = e & 1;
        int kk = ((j >> 2) << 4) + ((j & 3) << 1) + h + (hi << 3);
        af[e] = Qs[(wv << 4) + lh][kk];
    }
#pragma unroll
    for (int jt = 0; jt < 4; ++jt) {
        v16h bf;   // B(k=d, col=key): K^T -> read Ks[key][d]
#pragma unroll
        for (int e = 0; e < 16; ++e) bf[e] = Ks[(jt << 4) + lh][(hi << 3) + e + (e & 8)];
        v8f acc = {};
        acc = __builtin_amdgcn_wmma_f32_16x16x32_f16(false, af, false, bf, (short)0, acc, false, false);
#pragma unroll
        for (int r = 0; r < 8; ++r) {
            int qn = (wv << 4) + (hi << 3) + r, kn = (jt << 4) + lh;
            S[qn][kn] = acc[r] + bh[qn * 64 + kn];   // padding cols carry -1e30
        }
    }
    __syncthreads();

    // ---- softmax (one row per thread) ----
    if (tid < 64) {
        float m = -1e30f;
        for (int c = 0; c < 64; ++c) m = fmaxf(m, S[tid][c]);
        float s = 0.f;
        for (int c = 0; c < 64; ++c) { float e = __expf(S[tid][c] - m); s += e; S[tid][c] = e; }
        float inv = 1.0f / s;
        for (int c = 0; c < 64; ++c) P[tid][c] = (_Float16)(S[tid][c] * inv);
    }
    __syncthreads();

    // ---- P @ V (K = 64 keys, two K=32 steps) ----
    v8f acc2 = {};
#pragma unroll
    for (int ks = 0; ks < 2; ++ks) {
        v16h pa;
#pragma unroll
        for (int e = 0; e < 16; ++e) {
            int j = e >> 1, h = e & 1;
            int kk = ((j >> 2) << 4) + ((j & 3) << 1) + h + (hi << 3);
            pa[e] = P[(wv << 4) + lh][32 * ks + kk];
        }
        v16h vb;   // B(k=key, col=d): read Vs[key][d]
#pragma unroll
        for (int e = 0; e < 16; ++e) vb[e] = Vs[32 * ks + (hi << 3) + e + (e & 8)][lh];
        acc2 = __builtin_amdgcn_wmma_f32_16x16x32_f16(false, pa, false, vb, (short)0, acc2, false, false);
    }
#pragma unroll
    for (int r = 0; r < 8; ++r) {
        int qn = (wv << 4) + (hi << 3) + r;
        if (qn < 49) aout[((size_t)w * 49 + qn) * 128 + head * 16 + lh] = acc2[r];
    }
}

// ---------------- spatial gate * conv branch + BN -> concat right half (f16) ----------------
__global__ __launch_bounds__(128) void spatial_k(const float* __restrict__ aout,
    const float* __restrict__ cproj, const float* __restrict__ s1w, const float* __restrict__ s1b,
    const float* __restrict__ sbng, const float* __restrict__ sbnb,
    const float* __restrict__ s2w, const float* __restrict__ s2b,
    const float* __restrict__ cbg, const float* __restrict__ cbb, _Float16* __restrict__ conc)
{
    int p = blockIdx.x;                 // b*3136 + hw
    int b = p / 3136, hw = p % 3136;
    int y = hw / 56, x = hw % 56;
    int w = b * 64 + (y / 7) * 8 + (x / 7);
    int n = (y % 7) * 7 + (x % 7);
    __shared__ float ar[128], h[16], sg;
    int t = threadIdx.x;
    ar[t] = aout[((size_t)w * 49 + n) * 128 + t];
    __syncthreads();
    if (t < 16) {
        float s = s1b[t];
        for (int c = 0; c < 128; ++c) s += ar[c] * s1w[c * 16 + t];
        s = s * (sbng[t] * BN_RS) + sbnb[t];
        h[t] = gelu_f(s);
    }
    __syncthreads();
    if (t == 0) {
        float s = s2b[0];
        for (int j = 0; j < 16; ++j) s += h[j] * s2w[j];
        sg = sigmoid_f(s);
    }
    __syncthreads();
    float v = sg * cproj[(size_t)p * 128 + t];
    v = v * (cbg[t] * BN_RS) + cbb[t];
    conc[((size_t)w * 49 + n) * 256 + 128 + t] = (_Float16)v;
}

extern "C" void kernel_launch(void* const* d_in, const int* in_sizes, int n_in,
                              void* d_out, int out_size, void* d_ws, size_t ws_size,
                              hipStream_t stream) {
    (void)in_sizes; (void)n_in; (void)out_size; (void)ws_size;
    const float* x     = (const float*)d_in[0];
    const float* rpb   = (const float*)d_in[1];
    const float* paw   = (const float*)d_in[2];
    const float* pab   = (const float*)d_in[3];
    const float* alng  = (const float*)d_in[4];
    const float* alnb  = (const float*)d_in[5];
    const float* pcw   = (const float*)d_in[6];
    const float* pcb   = (const float*)d_in[7];
    const float* clng  = (const float*)d_in[8];
    const float* clnb  = (const float*)d_in[9];
    const float* dww   = (const float*)d_in[10];
    const float* dwb   = (const float*)d_in[11];
    const float* dwbng = (const float*)d_in[12];
    const float* dwbnb = (const float*)d_in[13];
    const float* ci1w  = (const float*)d_in[14];
    const float* ci1b  = (const float*)d_in[15];
    const float* cibng = (const float*)d_in[16];
    const float* cibnb = (const float*)d_in[17];
    const float* ci2w  = (const float*)d_in[18];
    const float* ci2b  = (const float*)d_in[19];
    const float* pjw   = (const float*)d_in[20];
    const float* pjb   = (const float*)d_in[21];
    const float* cbng  = (const float*)d_in[22];
    const float* cbnb  = (const float*)d_in[23];
    const float* qkvw  = (const float*)d_in[24];
    const float* qkvb  = (const float*)d_in[25];
    const float* si1w  = (const float*)d_in[26];
    const float* si1b  = (const float*)d_in[27];
    const float* sibng = (const float*)d_in[28];
    const float* sibnb = (const float*)d_in[29];
    const float* si2w  = (const float*)d_in[30];
    const float* si2b  = (const float*)d_in[31];
    const float* anlg  = (const float*)d_in[32];
    const float* anlb  = (const float*)d_in[33];
    const float* prw   = (const float*)d_in[34];
    const float* prb   = (const float*)d_in[35];

    char* ws = (char*)d_ws;
    _Float16* wh     = (_Float16*)(ws + OFF_W16);
    _Float16* paw_h  = wh;              // 256x128
    _Float16* pcw_h  = wh + 32768;      // 256x256
    _Float16* qkvw_h = wh + 98304;      // 128x384
    _Float16* pjw_h  = wh + 147456;     // 256x128
    _Float16* prw_h  = wh + 180224;     // 256x256
    float*    POOL   = (float*)(ws + OFF_POOL);
    float*    GATE   = (float*)(ws + OFF_GATE);
    float*    BIAS   = (float*)(ws + OFF_BIAS);
    float*    TCQKV  = (float*)(ws + OFF_R1);     // TC (Mx256) then QKV (Mx384)
    float*    XCNN   = (float*)(ws + OFF_R2);     // NCHW f32
    _Float16* CONCAT = (_Float16*)(ws + OFF_R2);  // later: Mx256 f16
    float*    CONV   = (float*)(ws + OFF_R3);     // NCHW f32
    float*    AOUT   = (float*)(ws + OFF_R3);     // later: Mx128 f32
    _Float16* XCH    = (_Float16*)(ws + OFF_R4);  // NHWC f16
    float*    TA     = (float*)(ws + OFF_R5);     // Mx128 f32
    float*    CPROJ  = (float*)(ws + OFF_R5);     // later: Mx128 f32
    _Float16* XATTH  = (_Float16*)(ws + OFF_R6);  // Mx128 f16

    // weights -> f16, padded attention bias table
    cvt_f16_k<<<128, 256, 0, stream>>>(paw,  paw_h,  32768);
    cvt_f16_k<<<256, 256, 0, stream>>>(pcw,  pcw_h,  65536);
    cvt_f16_k<<<192, 256, 0, stream>>>(qkvw, qkvw_h, 49152);
    cvt_f16_k<<<128, 256, 0, stream>>>(pjw,  pjw_h,  32768);
    cvt_f16_k<<<256, 256, 0, stream>>>(prw,  prw_h,  65536);
    bias_k<<<128, 256, 0, stream>>>(rpb, BIAS);

    // attention branch projection + LN
    gemm_wmma_k<float><<<dim3(MTOK / 64, 2), 128, 0, stream>>>(x, paw_h, pab, TA, MTOK, 256, 128);
    ln_k<<<MTOK / 4, 128, 0, stream>>>(TA, alng, alnb, nullptr, XATTH, 128, 128);

    // conv branch projection + LN + window_reverse + dwconv/BN/GELU
    gemm_wmma_k<float><<<dim3(MTOK / 64, 4), 128, 0, stream>>>(x, pcw_h, pcb, TCQKV, MTOK, 256, 256);
    ln_k<<<MTOK / 4, 128, 0, stream>>>(TCQKV, clng, clnb, TCQKV, nullptr, 256, 256);
    win_rev_k<<<100352, 256, 0, stream>>>(TCQKV, XCNN);
    dwconv_k<<<100352, 256, 0, stream>>>(XCNN, dww, dwb, dwbng, dwbnb, CONV);

    // channel interaction gate
    pool_k<<<8192, 256, 0, stream>>>(CONV, POOL);
    ci_k<<<32, 128, 0, stream>>>(POOL, ci1w, ci1b, cibng, cibnb, ci2w, ci2b, GATE);

    // conv branch 1x1 projection (NHWC GEMM)
    to_nhwc_h_k<<<100352, 256, 0, stream>>>(CONV, XCH);
    gemm_wmma_k<_Float16><<<dim3(MTOK / 64, 2), 128, 0, stream>>>(XCH, pjw_h, pjb, CPROJ, MTOK, 256, 128);

    // qkv + attention
    gemm_wmma_k<_Float16><<<dim3(MTOK / 64, 6), 128, 0, stream>>>(XATTH, qkvw_h, qkvb, TCQKV, MTOK, 128, 384);
    attn_k<<<16384, 128, 0, stream>>>(TCQKV, GATE, BIAS, AOUT);

    // spatial gate * conv branch + BN -> concat right half; LN(attn) -> left half
    spatial_k<<<100352, 128, 0, stream>>>(AOUT, CPROJ, si1w, si1b, sibng, sibnb,
                                          si2w, si2b, cbng, cbnb, CONCAT);
    ln_k<<<MTOK / 4, 128, 0, stream>>>(AOUT, anlg, anlb, nullptr, CONCAT, 128, 256);

    // final projection
    gemm_wmma_k<_Float16><<<dim3(MTOK / 64, 4), 128, 0, stream>>>(CONCAT, prw_h, prb,
                                                                  (float*)d_out, MTOK, 256, 256);
}
